// ConcatAttention_81853486727847
// MI455X (gfx1250) — compile-verified, matrix operand order checked by reading, not run on hardware
//
#include <hip/hip_runtime.h>
#include <hip/hip_bf16.h>
#include <math.h>

// Problem dims (fixed by the reference)
#define SS 2048
#define BB 32
#define EE 1024
#define HH 1024

typedef __attribute__((ext_vector_type(16))) _Float16 v16h_t;
typedef __attribute__((ext_vector_type(8)))  float    v8f_t;
typedef __attribute__((ext_vector_type(4)))  float    v4f_t;

// ---------------------------------------------------------------------------
// Kernel 1: v_enc[e] = sum_h w_score[h]*W_enc[h,e]   (stored as f16 for WMMA B)
//           v_dec[h] = sum_g w_score[g]*W_dec[g,h]   (stored as f32)
// blocks 0..3 -> v_enc, blocks 4..7 -> v_dec. 256 threads, coalesced columns.
// ---------------------------------------------------------------------------
__global__ void reduce_w_kernel(const float* __restrict__ W_enc,
                                const float* __restrict__ W_dec,
                                const float* __restrict__ w_score,
                                _Float16* __restrict__ venc16,
                                float* __restrict__ v_dec) {
    __shared__ float sw[HH];
    int tid = threadIdx.x;
    for (int i = tid; i < HH; i += 256) sw[i] = w_score[i];
    __syncthreads();

    if (blockIdx.x < 4) {
        int e = blockIdx.x * 256 + tid;
        float acc = 0.f;
        #pragma unroll 8
        for (int h = 0; h < HH; ++h) acc += sw[h] * W_enc[(size_t)h * EE + e];
        venc16[e] = (_Float16)acc;
    } else {
        int e = (blockIdx.x - 4) * 256 + tid;
        float acc = 0.f;
        #pragma unroll 8
        for (int h = 0; h < HH; ++h) acc += sw[h] * W_dec[(size_t)h * HH + e];
        v_dec[e] = acc;
    }
}

// ---------------------------------------------------------------------------
// Kernel 2: c[b] = dot(h_t[b,:], v_dec)   (32 blocks x 256 threads)
// ---------------------------------------------------------------------------
__global__ void ht_dot_kernel(const float* __restrict__ h_t,
                              const float* __restrict__ v_dec,
                              float* __restrict__ cb) {
    __shared__ float red[256];
    int b = blockIdx.x, tid = threadIdx.x;
    float acc = 0.f;
    for (int h = tid; h < HH; h += 256) acc += h_t[(size_t)b * HH + h] * v_dec[h];
    red[tid] = acc;
    __syncthreads();
    for (int off = 128; off > 0; off >>= 1) {
        if (tid < off) red[tid] += red[tid + off];
        __syncthreads();
    }
    if (tid == 0) cb[b] = red[0];
}

// ---------------------------------------------------------------------------
// Kernel 3: scores via WMMA. Flattened h_enc is a (S*B, E) matrix; row r=s*B+b.
// One wave per 16-row tile; K-loop of v_wmma_f32_16x16x32_f16 against a B
// fragment whose 16 columns all broadcast v_enc (f16, staged in LDS).
// D[m, n] == score(row m) for every n; lanes with N==0 extract, add c_b,
// tanh, and scatter into the alpha region of d_out at [b*S + s].
// 512 blocks x 256 threads (8 waves/block) = 4096 tiles = 65536 rows.
// ---------------------------------------------------------------------------
__global__ void scores_wmma_kernel(const float* __restrict__ h_enc,
                                   const _Float16* __restrict__ venc16,
                                   const float* __restrict__ cb,
                                   float* __restrict__ alpha_out) {
    __shared__ _Float16 sv[EE];
    int tid = threadIdx.x;
    for (int i = tid; i < EE; i += 256) sv[i] = venc16[i];
    __syncthreads();

    const int wave = tid >> 5;
    const int lane = tid & 31;
    const int hi   = lane >> 4;      // lane group (K-half select)
    const int hi8  = hi * 8;
    const int m    = lane & 15;      // row within tile
    const int tile = blockIdx.x * 8 + wave;
    const int rowBase = tile << 4;

    const float* rp = h_enc + (size_t)(rowBase + m) * EE;

    v8f_t acc = {};
    for (int kbase = 0; kbase < EE; kbase += 32) {
        // A fragment: 16-bit A-matrix 16x32 layout. Per lane:
        //   elements 0..7  <- K = kbase + hi*8 + [0..8)
        //   elements 8..15 <- K = kbase + 16 + hi*8 + [0..8)
        v4f_t p0 = *(const v4f_t*)(rp + kbase + hi8);
        v4f_t p1 = *(const v4f_t*)(rp + kbase + hi8 + 4);
        v4f_t p2 = *(const v4f_t*)(rp + kbase + 16 + hi8);
        v4f_t p3 = *(const v4f_t*)(rp + kbase + 16 + hi8 + 4);
        v16h_t a, bf;
        #pragma unroll
        for (int j = 0; j < 4; ++j) {
            a[j]      = (_Float16)p0[j];
            a[4 + j]  = (_Float16)p1[j];
            a[8 + j]  = (_Float16)p2[j];
            a[12 + j] = (_Float16)p3[j];
        }
        // B fragment: all 16 N-columns broadcast v_enc, so the value depends
        // only on K; use the mirrored K mapping.
        #pragma unroll
        for (int j = 0; j < 8; ++j) {
            bf[j]     = sv[kbase + hi8 + j];
            bf[8 + j] = sv[kbase + 16 + hi8 + j];
        }
        acc = __builtin_amdgcn_wmma_f32_16x16x32_f16(
            /*neg_a=*/false, a, /*neg_b=*/false, bf,
            /*c_mod=*/(short)0, acc, /*reuse_a=*/false, /*reuse_b=*/false);
    }

    // C/D layout: lane L (N = L&15), VGPR i -> M = i + (L<16 ? 0 : 8).
    if ((lane & 15) == 0) {
        #pragma unroll
        for (int i = 0; i < 8; ++i) {
            int r = rowBase + i + hi8;          // global row = s*B + b
            int b = r & (BB - 1);
            int s = r >> 5;
            float sc = tanhf(acc[i] + cb[b]);
            alpha_out[(size_t)b * SS + s] = sc;
        }
    }
}

// ---------------------------------------------------------------------------
// Kernel 4: softmax over S, in place on the alpha region (contiguous per b).
// 32 blocks x 256 threads, 8 elements/thread.
// ---------------------------------------------------------------------------
__global__ void softmax_kernel(float* __restrict__ alpha) {
    __shared__ float red[256];
    int b = blockIdx.x, tid = threadIdx.x;
    float* p = alpha + (size_t)b * SS;
    float x[8];
    float m = -INFINITY;
    #pragma unroll
    for (int j = 0; j < 8; ++j) { x[j] = p[tid + j * 256]; m = fmaxf(m, x[j]); }
    red[tid] = m; __syncthreads();
    for (int off = 128; off > 0; off >>= 1) {
        if (tid < off) red[tid] = fmaxf(red[tid], red[tid + off]);
        __syncthreads();
    }
    m = red[0]; __syncthreads();
    float s = 0.f;
    #pragma unroll
    for (int j = 0; j < 8; ++j) { x[j] = __expf(x[j] - m); s += x[j]; }
    red[tid] = s; __syncthreads();
    for (int off = 128; off > 0; off >>= 1) {
        if (tid < off) red[tid] += red[tid + off];
        __syncthreads();
    }
    float inv = 1.f / red[0];
    #pragma unroll
    for (int j = 0; j < 8; ++j) p[tid + j * 256] = x[j] * inv;
}

// ---------------------------------------------------------------------------
// Kernel 5: context[b,e] = sum_s alpha[b,s] * h_enc[s,b,e]. 128 blocks
// (4 e-chunks x 32 b) x 256 threads; alphas staged chunk-wise in LDS;
// prefetch the next s-chunk of h_enc.
// ---------------------------------------------------------------------------
__global__ void context_kernel(const float* __restrict__ h_enc,
                               const float* __restrict__ alpha,
                               float* __restrict__ ctx_out) {
    __shared__ float sA[256];
    int tid = threadIdx.x;
    int b = blockIdx.x >> 2;
    int e = ((blockIdx.x & 3) << 8) + tid;
    const float* p = h_enc + (size_t)b * EE + e;
    const float* ap = alpha + (size_t)b * SS;
    float acc = 0.f;
    for (int sc = 0; sc < SS; sc += 256) {
        sA[tid] = ap[sc + tid];
        __syncthreads();
        if (sc + 256 < SS)
            __builtin_prefetch(p + (size_t)(sc + 256) * (BB * EE), 0, 1);
        #pragma unroll 8
        for (int j = 0; j < 256; ++j)
            acc += sA[j] * p[(size_t)(sc + j) * (BB * EE)];
        __syncthreads();
    }
    ctx_out[(size_t)b * EE + e] = acc;
}

// ---------------------------------------------------------------------------
extern "C" void kernel_launch(void* const* d_in, const int* in_sizes, int n_in,
                              void* d_out, int out_size, void* d_ws, size_t ws_size,
                              hipStream_t stream) {
    (void)in_sizes; (void)n_in; (void)out_size; (void)ws_size;
    const float* h_t     = (const float*)d_in[0];   // (1,B,H)
    const float* h_enc   = (const float*)d_in[1];   // (S,B,E)
    const float* W_enc   = (const float*)d_in[2];   // (H,E)
    const float* W_dec   = (const float*)d_in[3];   // (H,H)
    const float* w_score = (const float*)d_in[4];   // (1,H)

    float* out = (float*)d_out;
    float* ctx_out   = out;                 // (1,B,E): B*E floats
    float* alpha_out = out + BB * EE;       // (B,S,1): B*S floats

    char* ws = (char*)d_ws;
    float*    v_dec  = (float*)ws;                  // 4 KB
    _Float16* venc16 = (_Float16*)(ws + 4096);      // 2 KB
    float*    cb     = (float*)(ws + 6144);         // 128 B

    reduce_w_kernel   <<<8,   256, 0, stream>>>(W_enc, W_dec, w_score, venc16, v_dec);
    ht_dot_kernel     <<<32,  256, 0, stream>>>(h_t, v_dec, cb);
    scores_wmma_kernel<<<512, 256, 0, stream>>>(h_enc, venc16, cb, alpha_out);
    softmax_kernel    <<<32,  256, 0, stream>>>(alpha_out);
    context_kernel    <<<128, 256, 0, stream>>>(h_enc, alpha_out, ctx_out);
}
